// image_to_word_85839216378365
// MI455X (gfx1250) — matrix operand order repaired and checked by reading, not compiled, not run on hardware
//
#include <hip/hip_runtime.h>
#include <hip/hip_bf16.h>
#include <math.h>

// MI455X / gfx1250 implementation: ResNet-18 (bf16 WMMA implicit GEMM via im2col,
// NHWC activations, f32 accumulate, BN folded into GEMM epilogue) + serial GRU
// (bf16 WMMA GEMMs + fused f32 gate kernel) + WMMA projection.
// Each wave computes a 16x32 C tile (two v_wmma_f32_16x16x32_bf16 per K-step
// sharing one A fragment) -> 3 b128 loads per WMMA instead of 4.

typedef __attribute__((ext_vector_type(16))) __bf16 v16bf;
typedef __attribute__((ext_vector_type(8)))  __bf16 v8bf;
typedef __attribute__((ext_vector_type(8)))  float  v8f;

#define NB   64
#define DHID 512
#define NWRD 8000
#define TMAX 38
#define CDIV(a,b) (((a)+(b)-1)/(b))

__device__ __forceinline__ float  bf2f(__bf16 x) { return (float)x; }
__device__ __forceinline__ __bf16 f2bf(float x)  { return (__bf16)x; }

// ---------------- WMMA 16x32 tile accumulation over K ----------------
// A: [16, lda] row-major bf16 tile (rows = M).  B0/B1: two adjacent [16, ldb]
// row-major bf16 tiles where rows = N (B stored transposed: each row is a
// K-vector).  This code uses contiguous k order, the same for A and B:
//   A lane l: m = l&15; contiguous 16-K run at (l>=16 ? 16 : 0)
//   B lane l: n = l&15; contiguous 16-K run at (l>=16 ? 16 : 0)
__device__ __forceinline__ void wmma_tile2(const __bf16* __restrict__ A, long lda,
                                           const __bf16* __restrict__ B, long ldb,
                                           int K, v8f& c0, v8f& c1)
{
  const int l    = threadIdx.x & 31;
  const int half = l >> 4;
  const int idx  = l & 15;
  const __bf16* pa  = A + (long)idx * lda + half * 16;
  const __bf16* pb0 = B + (long)idx * ldb + half * 16;
  const __bf16* pb1 = pb0 + 16 * ldb;
  for (int k = 0; k < K; k += 32) {
    v8bf a0 = *(const v8bf*)(pa);
    v8bf a1 = *(const v8bf*)(pa + 8);
    v16bf av = __builtin_shufflevector(a0, a1,
        0,1,2,3,4,5,6,7,8,9,10,11,12,13,14,15);
    v16bf bv0 = *(const v16bf*)(pb0);
    v16bf bv1 = *(const v16bf*)(pb1);
    c0 = __builtin_amdgcn_wmma_f32_16x16x32_bf16(false, av, false, bv0,
                                                 (short)0, c0, false, false);
    c1 = __builtin_amdgcn_wmma_f32_16x16x32_bf16(false, av, false, bv1,
                                                 (short)0, c1, false, false);
    pa += 32; pb0 += 32; pb1 += 32;
  }
}

// ---------------- GEMM with conv epilogue (scale/bias/residual/relu -> bf16) --
// Block tile 64(M) x 64(N): 8 waves = 4 M-groups x 2 N-pairs, wave tile 16x32.
// Requires M%64==0, N%64==0, K%32==0 (guaranteed) -> no divergence, EXEC all-1s.
__global__ void __launch_bounds__(256)
gemm_epi_bf16_kernel(const __bf16* __restrict__ A, const __bf16* __restrict__ B,
                     __bf16* __restrict__ out, long M, long N, long K,
                     const float* __restrict__ scale,
                     const float* __restrict__ bias,
                     const __bf16* __restrict__ res, int relu)
{
  const int w  = threadIdx.x >> 5;
  const long m0 = (long)blockIdx.y * 64 + (long)(w >> 1) * 16;
  const long n0 = (long)blockIdx.x * 64 + (long)(w & 1) * 32;
  v8f c0 = {}, c1 = {};
  wmma_tile2(A + m0 * K, K, B + n0 * K, K, (int)K, c0, c1);
  const int l    = threadIdx.x & 31;
  const int half = l >> 4;
  const long n   = n0 + (l & 15);
  const float s0 = scale[n],      b0 = bias[n];
  const float s1 = scale[n + 16], b1 = bias[n + 16];
#pragma unroll
  for (int r = 0; r < 8; ++r) {
    long m  = m0 + r + 8 * half;
    float v0 = c0[r] * s0 + b0;
    float v1 = c1[r] * s1 + b1;
    if (res) {
      v0 += bf2f(res[m * N + n]);
      v1 += bf2f(res[m * N + n + 16]);
    }
    if (relu) { v0 = fmaxf(v0, 0.f); v1 = fmaxf(v1, 0.f); }
    out[m * N + n]      = f2bf(v0);
    out[m * N + n + 16] = f2bf(v1);
  }
}

// ---------------- GEMM, plain fp32 output ----------------
__global__ void __launch_bounds__(256)
gemm_f32_kernel(const __bf16* __restrict__ A, const __bf16* __restrict__ B,
                float* __restrict__ C, long M, long N, long K)
{
  const int w  = threadIdx.x >> 5;
  const long m0 = (long)blockIdx.y * 64 + (long)(w >> 1) * 16;
  const long n0 = (long)blockIdx.x * 64 + (long)(w & 1) * 32;
  v8f c0 = {}, c1 = {};
  wmma_tile2(A + m0 * K, K, B + n0 * K, K, (int)K, c0, c1);
  const int l    = threadIdx.x & 31;
  const int half = l >> 4;
  const long n   = n0 + (l & 15);
#pragma unroll
  for (int r = 0; r < 8; ++r) {
    long m = m0 + r + 8 * half;
    C[m * N + n]      = c0[r];
    C[m * N + n + 16] = c1[r];
  }
}

// ---------------- weight repack: OIHW fp32 -> [Cout, (kh,kw,ci)] bf16, K padded
__global__ void wprep_kernel(const float* __restrict__ w, __bf16* __restrict__ dst,
                             int Cout, int Cin, int KH, int K, int Kpad)
{
  long idx = (long)blockIdx.x * blockDim.x + threadIdx.x;
  long total = (long)Cout * Kpad;
  if (idx >= total) return;
  int k  = (int)(idx % Kpad);
  int co = (int)(idx / Kpad);
  if (k >= K) { dst[idx] = f2bf(0.f); return; }
  int ci = k % Cin, p = k / Cin;
  int kw = p % KH,  kh = p / KH;
  dst[idx] = f2bf(w[(((long)co * Cin + ci) * KH + kh) * KH + kw]);
}

__global__ void cvt_kernel(const float* __restrict__ src, __bf16* __restrict__ dst, long n)
{
  long idx = (long)blockIdx.x * blockDim.x + threadIdx.x;
  if (idx < n) dst[idx] = f2bf(src[idx]);
}

// ---------------- im2col: stem (NCHW fp32 input image) ----------------
// K = 3*7*7 = 147 (order (kh,kw,ci)), Kpad = 160, stride 2, pad 3, 224 -> 112
__global__ void im2col_stem_kernel(const float* __restrict__ img,
                                   __bf16* __restrict__ dst, int Kpad)
{
  long idx = (long)blockIdx.x * blockDim.x + threadIdx.x;
  long total = (long)NB * 112 * 112 * Kpad;
  if (idx >= total) return;
  int k = (int)(idx % Kpad);
  long row = idx / Kpad;
  if (k >= 147) { dst[idx] = f2bf(0.f); return; }
  int wo = (int)(row % 112); long r2 = row / 112;
  int ho = (int)(r2 % 112);  int n  = (int)(r2 / 112);
  int ci = k % 3, p = k / 3;
  int kw = p % 7, kh = p / 7;
  int hi = ho * 2 + kh - 3, wi = wo * 2 + kw - 3;
  float v = 0.f;
  if (hi >= 0 && hi < 224 && wi >= 0 && wi < 224)
    v = img[(((long)n * 3 + ci) * 224 + hi) * 224 + wi];
  dst[idx] = f2bf(v);
}

// ---------------- im2col: general NHWC bf16 source (square images) ----------
__global__ void im2col_nhwc_kernel(const __bf16* __restrict__ src,
                                   __bf16* __restrict__ dst,
                                   int Cin, int Hin, int KH, int st, int pad,
                                   int Hout, int K, int Kpad)
{
  long idx = (long)blockIdx.x * blockDim.x + threadIdx.x;
  long total = (long)NB * Hout * Hout * Kpad;
  if (idx >= total) return;
  int k = (int)(idx % Kpad);
  long row = idx / Kpad;
  if (k >= K) { dst[idx] = f2bf(0.f); return; }
  int wo = (int)(row % Hout); long r2 = row / Hout;
  int ho = (int)(r2 % Hout);  int n  = (int)(r2 / Hout);
  int ci = k % Cin, p = k / Cin;
  int kw = p % KH,  kh = p / KH;
  int hi = ho * st + kh - pad, wi = wo * st + kw - pad;
  __bf16 v = f2bf(0.f);
  if (hi >= 0 && hi < Hin && wi >= 0 && wi < Hin)
    v = src[(((long)n * Hin + hi) * Hin + wi) * Cin + ci];
  dst[idx] = v;
}

// ---------------- maxpool 3x3 s2 pad1, NHWC bf16 ----------------
__global__ void maxpool_kernel(const __bf16* __restrict__ src, __bf16* __restrict__ dst,
                               int C, int Hin, int Hout)
{
  long idx = (long)blockIdx.x * blockDim.x + threadIdx.x;
  long total = (long)NB * Hout * Hout * C;
  if (idx >= total) return;
  int c = (int)(idx % C); long r = idx / C;
  int wo = (int)(r % Hout); r /= Hout;
  int ho = (int)(r % Hout); int n = (int)(r / Hout);
  float m = -INFINITY;
  for (int kh = 0; kh < 3; ++kh)
    for (int kw = 0; kw < 3; ++kw) {
      int hi = 2 * ho + kh - 1, wi = 2 * wo + kw - 1;
      if (hi >= 0 && hi < Hin && wi >= 0 && wi < Hin)
        m = fmaxf(m, bf2f(src[(((long)n * Hin + hi) * Hin + wi) * C + c]));
    }
  dst[idx] = f2bf(m);
}

// ---------------- global avg pool 7x7, NHWC [64,7,7,512] -> h (f32 + bf16) ---
__global__ void avgpool_kernel(const __bf16* __restrict__ src,
                               float* __restrict__ hf, __bf16* __restrict__ hb)
{
  int idx = blockIdx.x * blockDim.x + threadIdx.x;
  if (idx >= NB * DHID) return;
  int b = idx / DHID, c = idx % DHID;
  float s = 0.f;
  for (int h = 0; h < 7; ++h)
    for (int w = 0; w < 7; ++w)
      s += bf2f(src[(((long)b * 7 + h) * 7 + w) * DHID + c]);
  float m = s * (1.f / 49.f);
  hf[idx] = m;
  hb[idx] = f2bf(m);
}

// ---------------- SOS row: inp[0][b][:] = embed[0][:] ----------------
__global__ void sos_kernel(const float* __restrict__ embed, __bf16* __restrict__ inp)
{
  int idx = blockIdx.x * blockDim.x + threadIdx.x;
  if (idx >= NB * DHID) return;
  inp[idx] = f2bf(embed[idx % DHID]);
}

__global__ void copyrow_kernel(const __bf16* __restrict__ src, __bf16* __restrict__ dst)
{
  int idx = blockIdx.x * blockDim.x + threadIdx.x;
  if (idx < NB * DHID) dst[idx] = src[idx];
}

// ---------------- fused GRU gates (PyTorch r,z,n layout) ----------------
__global__ void gru_gate_kernel(const float* __restrict__ GX, const float* __restrict__ GH,
                                const float* __restrict__ bi, const float* __restrict__ bh,
                                float* __restrict__ h, __bf16* __restrict__ hb)
{
  int idx = blockIdx.x * blockDim.x + threadIdx.x;
  if (idx >= NB * DHID) return;
  int b = idx >> 9, d = idx & 511;
  const float* gx = GX + (long)b * 1536;
  const float* gh = GH + (long)b * 1536;
  float rx = gx[d]        + bi[d],        rh = gh[d]        + bh[d];
  float zx = gx[512 + d]  + bi[512 + d],  zh = gh[512 + d]  + bh[512 + d];
  float nx = gx[1024 + d] + bi[1024 + d], nh = gh[1024 + d] + bh[1024 + d];
  float r  = 1.f / (1.f + expf(-(rx + rh)));
  float z  = 1.f / (1.f + expf(-(zx + zh)));
  float n  = tanhf(nx + r * nh);
  float hv = (1.f - z) * n + z * h[idx];
  h[idx]  = hv;
  hb[idx] = f2bf(hv);
}

// ---------------- projection epilogue: [t*64+b, v] -> out[b, v, t] + bp[v] ---
__global__ void proj_out_kernel(const float* __restrict__ C, const float* __restrict__ bp,
                                float* __restrict__ out)
{
  long idx = (long)blockIdx.x * blockDim.x + threadIdx.x;
  long total = (long)TMAX * NB * NWRD;
  if (idx >= total) return;
  int v = (int)(idx % NWRD);
  long row = idx / NWRD;
  int b = (int)(row % NB);
  int t = (int)(row / NB);
  out[((long)b * NWRD + v) * TMAX + t] = C[idx] + bp[v];
}

// =====================================================================
extern "C" void kernel_launch(void* const* d_in, const int* in_sizes, int n_in,
                              void* d_out, int out_size, void* d_ws, size_t ws_size,
                              hipStream_t stream)
{
  (void)in_sizes; (void)n_in; (void)out_size; (void)ws_size;
  const float* img = (const float*)d_in[0];

  // ---- flattened input index map (setup_inputs() dict order) ----
  const bool hasD[8] = {false,false,true,false,true,false,true,false};
  int iw1[8], is1[8], ib1[8], iw2[8], is2[8], ib2[8], iwd[8], isd[8], ibd[8];
  int id = 4;  // 0=img, 1=stem_w, 2=stem_s, 3=stem_b
  for (int i = 0; i < 8; ++i) {
    iw1[i]=id; is1[i]=id+1; ib1[i]=id+2; iw2[i]=id+3; is2[i]=id+4; ib2[i]=id+5; id += 6;
    if (hasD[i]) { iwd[i]=id; isd[i]=id+1; ibd[i]=id+2; id += 3; }
    else         { iwd[i]=isd[i]=ibd[i]=-1; }
  }
  const int iWi = id, iWh = id+1, ibi = id+2, ibh = id+3,
            iEmb = id+4, iWp = id+5, ibp = id+6;

  // ---- conv layer table ----
  struct Conv { int wi,si,bi, Cin,Cout,KH,st,pad, Hin,Hout, K; long Kpad; size_t woff; };
  Conv cv[17]; int nc = 0;
  size_t woff = 0;
  auto addc = [&](int wi,int si,int bi,int Cin,int Cout,int KH,int st,int pad,
                  int Hin,int Hout)->int {
    Conv& c = cv[nc];
    c.wi=wi; c.si=si; c.bi=bi; c.Cin=Cin; c.Cout=Cout; c.KH=KH; c.st=st; c.pad=pad;
    c.Hin=Hin; c.Hout=Hout;
    c.K = Cin * KH * KH;
    c.Kpad = (long)((c.K + 31) & ~31);
    c.woff = woff;
    woff += (size_t)Cout * c.Kpad;
    return nc++;
  };
  const int Lstem = addc(1,2,3, 3,64,7,2,3, 224,112);
  struct BG { int cin,cout,st,hin,hout; };
  const BG bg[8] = {{64,64,1,56,56},{64,64,1,56,56},
                    {64,128,2,56,28},{128,128,1,28,28},
                    {128,256,2,28,14},{256,256,1,14,14},
                    {256,512,2,14,7},{512,512,1,7,7}};
  int Lc1[8], Lc2[8], Ld[8];
  for (int i = 0; i < 8; ++i) {
    Lc1[i] = addc(iw1[i],is1[i],ib1[i], bg[i].cin, bg[i].cout, 3, bg[i].st, 1,
                  bg[i].hin, bg[i].hout);
    Ld[i]  = hasD[i] ? addc(iwd[i],isd[i],ibd[i], bg[i].cin, bg[i].cout, 1,
                            bg[i].st, 0, bg[i].hin, bg[i].hout) : -1;
    Lc2[i] = addc(iw2[i],is2[i],ib2[i], bg[i].cout, bg[i].cout, 3, 1, 1,
                  bg[i].hout, bg[i].hout);
  }
  const size_t offWi = woff;             woff += 1536L * 512;
  const size_t offWh = woff;             woff += 1536L * 512;
  const size_t offWp = woff;             woff += (size_t)NWRD * 512;

  // ---- workspace carve ----
  char* ws = (char*)d_ws; size_t off = 0;
  auto take = [&](size_t bytes)->void* {
    void* p = ws + off; off = (off + bytes + 255) & ~(size_t)255; return p;
  };
  __bf16* wbuf = (__bf16*)take(woff * sizeof(__bf16));
  size_t colMax = 0;
  for (int i = 0; i < nc; ++i) {
    size_t e = (size_t)NB * cv[i].Hout * cv[i].Hout * cv[i].Kpad;
    if (e > colMax) colMax = e;
  }
  __bf16* colbuf = (__bf16*)take(colMax * sizeof(__bf16));
  const size_t actE = (size_t)NB * 112 * 112 * 64;
  __bf16* pA = (__bf16*)take(actE * sizeof(__bf16));
  __bf16* pB = (__bf16*)take(actE * sizeof(__bf16));
  __bf16* pC = (__bf16*)take(actE * sizeof(__bf16));
  float*  GX   = (float*)take((size_t)TMAX * NB * 1536 * sizeof(float));
  float*  GH   = (float*)take((size_t)NB * 1536 * sizeof(float));
  float*  hF   = (float*)take((size_t)NB * DHID * sizeof(float));
  __bf16* hB   = (__bf16*)take((size_t)NB * DHID * sizeof(__bf16));
  __bf16* inpB = (__bf16*)take((size_t)TMAX * NB * DHID * sizeof(__bf16));
  float*  projC = (float*)take((size_t)TMAX * NB * NWRD * sizeof(float));

  // ---- weight prep (repack/convert to bf16, K-padded) ----
  for (int i = 0; i < nc; ++i) {
    Conv& c = cv[i];
    long tot = (long)c.Cout * c.Kpad;
    wprep_kernel<<<CDIV(tot,256),256,0,stream>>>((const float*)d_in[c.wi],
        wbuf + c.woff, c.Cout, c.Cin, c.KH, c.K, (int)c.Kpad);
  }
  cvt_kernel<<<CDIV(1536L*512,256),256,0,stream>>>((const float*)d_in[iWi], wbuf+offWi, 1536L*512);
  cvt_kernel<<<CDIV(1536L*512,256),256,0,stream>>>((const float*)d_in[iWh], wbuf+offWh, 1536L*512);
  cvt_kernel<<<CDIV((long)NWRD*512,256),256,0,stream>>>((const float*)d_in[iWp], wbuf+offWp, (long)NWRD*512);

  // ---- conv helper ----
  auto conv_fwd = [&](int li, const __bf16* src, const __bf16* res, int relu,
                      __bf16* out) {
    Conv& c = cv[li];
    long M = (long)NB * c.Hout * c.Hout;
    long tot = M * c.Kpad;
    im2col_nhwc_kernel<<<CDIV(tot,256),256,0,stream>>>(src, colbuf,
        c.Cin, c.Hin, c.KH, c.st, c.pad, c.Hout, c.K, (int)c.Kpad);
    dim3 g((unsigned)(c.Cout / 64), (unsigned)(M / 64));
    gemm_epi_bf16_kernel<<<g,256,0,stream>>>(colbuf, wbuf + c.woff, out,
        M, (long)c.Cout, c.Kpad,
        (const float*)d_in[c.si], (const float*)d_in[c.bi], res, relu);
  };

  // ---- stem: conv7x7 s2 + BN + relu, then maxpool ----
  {
    Conv& c = cv[Lstem];
    long M = (long)NB * 112 * 112;
    long tot = M * c.Kpad;
    im2col_stem_kernel<<<CDIV(tot,256),256,0,stream>>>(img, colbuf, (int)c.Kpad);
    dim3 g(1, (unsigned)(M / 64));
    gemm_epi_bf16_kernel<<<g,256,0,stream>>>(colbuf, wbuf + c.woff, pA,
        M, 64L, c.Kpad, (const float*)d_in[2], (const float*)d_in[3], nullptr, 1);
  }
  {
    long tot = (long)NB * 56 * 56 * 64;
    maxpool_kernel<<<CDIV(tot,256),256,0,stream>>>(pA, pB, 64, 112, 56);
  }

  // ---- residual blocks (NHWC bf16 activations) ----
  __bf16 *X = pB, *T1 = pA, *T2 = pC;
  for (int i = 0; i < 8; ++i) {
    conv_fwd(Lc1[i], X, nullptr, 1, T1);            // y1 = relu(bn(conv1(x)))
    const __bf16* res; __bf16* out;
    if (hasD[i]) {
      conv_fwd(Ld[i], X, nullptr, 0, T2);           // sc = bn(conv_d(x))
      res = T2; out = X;                            // x fully consumed by im2cols
    } else {
      res = X;  out = T2;
    }
    conv_fwd(Lc2[i], T1, res, 1, out);              // out = relu(bn(conv2(y1)) + sc)
    if (!hasD[i]) { __bf16* t = X; X = T2; T2 = t; }
    // hasD: out == X already
  }

  // ---- avg pool -> initial hidden (f32 + bf16) ----
  avgpool_kernel<<<CDIV(NB*DHID,256),256,0,stream>>>(X, hF, hB);

  // ---- GRU (faithful to the buggy growing-sequence loop) ----
  sos_kernel<<<CDIV(NB*DHID,256),256,0,stream>>>((const float*)d_in[iEmb], inpB);
  auto gemm32 = [&](const __bf16* A, const __bf16* B, float* C,
                    long M, long N, long K) {
    dim3 g((unsigned)(N / 64), (unsigned)(M / 64));
    gemm_f32_kernel<<<g,256,0,stream>>>(A, B, C, M, N, K);
  };
  gemm32(inpB, wbuf + offWi, GX, 64, 1536, 512);     // GX row 0 (SOS)
  for (int L = 1; L <= TMAX - 1; ++L) {
    for (int t = 0; t < L; ++t) {
      gemm32(hB, wbuf + offWh, GH, 64, 1536, 512);   // h @ Wh^T
      gru_gate_kernel<<<CDIV(NB*DHID,256),256,0,stream>>>(
          GX + (long)t * NB * 1536, GH,
          (const float*)d_in[ibi], (const float*)d_in[ibh], hF, hB);
    }
    copyrow_kernel<<<CDIV(NB*DHID,256),256,0,stream>>>(hB, inpB + (long)L*NB*DHID);
    gemm32(inpB + (long)L*NB*DHID, wbuf + offWi,
           GX + (long)L*NB*1536, 64, 1536, 512);     // gates for the new row
  }

  // ---- projection: [38*64,512] x [8000,512]^T, then bias + permute ----
  gemm32(inpB, wbuf + offWp, projC, (long)TMAX * NB, NWRD, 512);
  {
    long tot = (long)TMAX * NB * NWRD;
    proj_out_kernel<<<CDIV(tot,256),256,0,stream>>>(projC,
        (const float*)d_in[ibp], (float*)d_out);
  }
}
